// Attention_76802605187492
// MI455X (gfx1250) — matrix operand order repaired and checked by reading, not compiled, not run on hardware
//
#include <hip/hip_runtime.h>
#include <hip/hip_bf16.h>

// MI455X / gfx1250 implementation.
//
// Shapes: B=64, T=512, H=1024, A=300.  Dominant cost: hidden@W_h.T
// (32768x1024x1024 = 68.7 GFLOP) fused directly into attention scores so the
// [B,T,H] activation is never written to memory.  The aspect branch
// contributes a per-batch-constant to pre-softmax scores and is removed by
// softmax shift-invariance.  Matmuls run on v_wmma_f32_16x16x32_f16 (f32
// accumulate).  tanh uses the CDNA5 hardware v_tanh_f32 when available.

typedef __attribute__((ext_vector_type(16))) _Float16 v16h;
typedef __attribute__((ext_vector_type(8)))  _Float16 v8h;
typedef __attribute__((ext_vector_type(4)))  _Float16 v4h;
typedef __attribute__((ext_vector_type(8)))  float    v8f;
typedef __attribute__((ext_vector_type(4)))  float    v4f;

#define Bsz   64
#define Tseq  512
#define Hdim  1024
#define Mrows (Bsz * Tseq)

// Fast tanh: prefer the gfx1250 hardware transcendental; otherwise a branch-
// free rational form on the fast hardware exp (clamped so exp never overflows).
__device__ __forceinline__ float fast_tanh(float x) {
#if __has_builtin(__builtin_amdgcn_tanhf)
  return __builtin_amdgcn_tanhf(x);
#else
  x = fminf(fmaxf(x, -15.f), 15.f);
  float e = __expf(2.f * x);
  return (e - 1.f) * __builtin_amdgcn_rcpf(e + 1.f);
#endif
}

// ---------------------------------------------------------------- utilities
__global__ __launch_bounds__(256)
void cvt_f32_f16_kernel(const float* __restrict__ src, _Float16* __restrict__ dst, int n) {
  int i = blockIdx.x * 256 + threadIdx.x;
  if (i < n) dst[i] = (_Float16)src[i];
}

// --------------------------------------------------- fused GEMM -> scores
// scores[m] = sum_n tanh( (hidden @ W_h.T)[m,n] + b_h[n] ) * w_w[n]
// One block = 32 rows of M (two 16-row A-tiles in LDS).  8 waves; wave w owns
// N-tiles {w, w+8, ..., w+56}; each B fragment is reused by two WMMAs.
__global__ __launch_bounds__(256)
void scores_kernel(const float* __restrict__ hidden,
                   const _Float16* __restrict__ wh16,   // W_h in f16, [H,H] row-major
                   const float* __restrict__ b_h,
                   const float* __restrict__ w_w,
                   float* __restrict__ scores) {
  __shared__ _Float16 lds_a[32 * Hdim];   // 64 KB: two 16x1024 f16 A-tiles
  __shared__ float    sscore[32];

  const int tid = threadIdx.x;
  const int m0  = blockIdx.x * 32;

  // Stage 32x1024 hidden rows into LDS as f16 (vectorized f32x4 -> f16x4).
  #pragma unroll
  for (int i = 0; i < 32; ++i) {
    int f4  = tid + 256 * i;          // 8192 float4 chunks
    int row = f4 >> 8;
    int k   = (f4 & 255) << 2;
    v4f v = *(const v4f*)(hidden + (size_t)(m0 + row) * Hdim + k);
    v4h h; h[0] = (_Float16)v[0]; h[1] = (_Float16)v[1];
           h[2] = (_Float16)v[2]; h[3] = (_Float16)v[3];
    *(v4h*)(lds_a + row * Hdim + k) = h;
  }
  if (tid < 32) sscore[tid] = 0.f;
  __syncthreads();

  const int wave  = tid >> 5;
  const int lane  = tid & 31;
  const int half  = lane >> 4;    // which 16-lane half of the wave
  const int l16   = lane & 15;
  const int akoff = half * 16;    // A fragment: lanes 16-31 take upper K-half
  const int bkoff = half * 16;    // B fragment: lanes 16-31 take upper K-half

  float ps0[8], ps1[8];
  #pragma unroll
  for (int r = 0; r < 8; ++r) { ps0[r] = 0.f; ps1[r] = 0.f; }

  const _Float16* arow0 = lds_a + l16 * Hdim;           // A-tile 0, this lane's row
  const _Float16* arow1 = lds_a + (16 + l16) * Hdim;    // A-tile 1

  for (int i = 0; i < 8; ++i) {
    const int nt = wave + 8 * i;
    const int n  = nt * 16 + l16;                  // this lane's output column
    const _Float16* brow = wh16 + (size_t)n * Hdim;

    v8f acc0 = {}, acc1 = {};
    #pragma unroll 2
    for (int kt = 0; kt < Hdim; kt += 32) {
      // B (32x16 f16): B[k,n] = W_h[n,k] -> 16 contiguous halfs from row n
      v16h bv = *(const v16h*)(brow + kt + bkoff);
      // A (16x32 f16): lane row = l16; K = kt+akoff+{0..15}
      v8h a0lo = *(const v8h*)(arow0 + kt + akoff);
      v8h a0hi = *(const v8h*)(arow0 + kt + akoff + 8);
      v8h a1lo = *(const v8h*)(arow1 + kt + akoff);
      v8h a1hi = *(const v8h*)(arow1 + kt + akoff + 8);
      v16h a0 = __builtin_shufflevector(a0lo, a0hi,
                  0,1,2,3,4,5,6,7,8,9,10,11,12,13,14,15);
      v16h a1 = __builtin_shufflevector(a1lo, a1hi,
                  0,1,2,3,4,5,6,7,8,9,10,11,12,13,14,15);
      acc0 = __builtin_amdgcn_wmma_f32_16x16x32_f16(
               false, a0, false, bv, (short)0, acc0, false, false);
      acc1 = __builtin_amdgcn_wmma_f32_16x16x32_f16(
               false, a1, false, bv, (short)0, acc1, false, false);
    }

    const float bias = b_h[n];
    const float ww   = w_w[n];
    #pragma unroll
    for (int r = 0; r < 8; ++r) {
      ps0[r] += fast_tanh(acc0[r] + bias) * ww;  // D[r]: row (r+8*half), col n
      ps1[r] += fast_tanh(acc1[r] + bias) * ww;
    }
  }

  // Reduce over the 16 lanes of each half-wave (same M row, different N cols).
  #pragma unroll
  for (int r = 0; r < 8; ++r) {
    float v0 = ps0[r], v1 = ps1[r];
    v0 += __shfl_xor(v0, 1, 32);  v1 += __shfl_xor(v1, 1, 32);
    v0 += __shfl_xor(v0, 2, 32);  v1 += __shfl_xor(v1, 2, 32);
    v0 += __shfl_xor(v0, 4, 32);  v1 += __shfl_xor(v1, 4, 32);
    v0 += __shfl_xor(v0, 8, 32);  v1 += __shfl_xor(v1, 8, 32);
    if (l16 == 0) {
      atomicAdd(&sscore[r + 8 * half],      v0);
      atomicAdd(&sscore[16 + r + 8 * half], v1);
    }
  }
  __syncthreads();
  if (tid < 32) scores[m0 + tid] = sscore[tid];
}

// ------------------------------------------------------------ softmax over T
__global__ __launch_bounds__(512)
void softmax_kernel(const float* __restrict__ scores, float* __restrict__ alpha) {
  __shared__ float red[512];
  const int b = blockIdx.x, t = threadIdx.x;
  float s = scores[b * Tseq + t];
  red[t] = s;
  __syncthreads();
  for (int off = 256; off > 0; off >>= 1) {
    if (t < off) red[t] = fmaxf(red[t], red[t + off]);
    __syncthreads();
  }
  float mx = red[0];
  __syncthreads();
  float e = __expf(s - mx);
  red[t] = e;
  __syncthreads();
  for (int off = 256; off > 0; off >>= 1) {
    if (t < off) red[t] += red[t + off];
    __syncthreads();
  }
  alpha[b * Tseq + t] = e / red[0];
}

// --------------------------------------------------- r[b,h] = sum_t a*hidden
__global__ __launch_bounds__(256)
void weighted_sum_kernel(const float* __restrict__ hidden,
                         const float* __restrict__ alpha,
                         float* __restrict__ r) {
  const int b = blockIdx.y;
  const int h = blockIdx.x * 256 + threadIdx.x;
  const float* hb = hidden + (size_t)b * Tseq * Hdim + h;
  const float* ab = alpha + b * Tseq;          // uniform -> scalar loads
  float acc = 0.f;
  #pragma unroll 8
  for (int t = 0; t < Tseq; ++t)
    acc = fmaf(ab[t], hb[(size_t)t * Hdim], acc);
  r[b * Hdim + h] = acc;
}

// ---------------------------------- 64xK GEMM via WMMA: out = A@W.T + bias
// grid = (4 m-tiles, 8 n-groups), 8 waves/block, wave -> one 16-wide N tile.
__global__ __launch_bounds__(256)
void gemm64_wmma_kernel(const float* __restrict__ A, long long arowstride,
                        const _Float16* __restrict__ B16,
                        const float* __restrict__ bias,
                        float* __restrict__ out) {
  __shared__ _Float16 lds_a[16 * Hdim];
  const int tid = threadIdx.x;
  const int m0  = blockIdx.x * 16;

  #pragma unroll
  for (int i = 0; i < 16; ++i) {
    int f4  = tid + 256 * i;
    int row = f4 >> 8;
    int k   = (f4 & 255) << 2;
    v4f v = *(const v4f*)(A + (long long)(m0 + row) * arowstride + k);
    v4h h; h[0] = (_Float16)v[0]; h[1] = (_Float16)v[1];
           h[2] = (_Float16)v[2]; h[3] = (_Float16)v[3];
    *(v4h*)(lds_a + row * Hdim + k) = h;
  }
  __syncthreads();

  const int wave = tid >> 5, lane = tid & 31;
  const int half = lane >> 4, l16 = lane & 15;
  const int akoff = half * 16, bkoff = half * 16;
  const int nt = blockIdx.y * 8 + wave;
  const int n  = nt * 16 + l16;
  const _Float16* brow = B16 + (size_t)n * Hdim;

  v8f acc = {};
  #pragma unroll 4
  for (int kt = 0; kt < Hdim; kt += 32) {
    v8h alo = *(const v8h*)(lds_a + l16 * Hdim + kt + akoff);
    v8h ahi = *(const v8h*)(lds_a + l16 * Hdim + kt + akoff + 8);
    v16h av = __builtin_shufflevector(alo, ahi,
                0,1,2,3,4,5,6,7,8,9,10,11,12,13,14,15);
    v16h bv = *(const v16h*)(brow + kt + bkoff);
    acc = __builtin_amdgcn_wmma_f32_16x16x32_f16(
            false, av, false, bv, (short)0, acc, false, false);
  }
  const float bb = bias[n];
  #pragma unroll
  for (int rr = 0; rr < 8; ++rr)
    out[(size_t)(m0 + rr + 8 * half) * Hdim + n] = acc[rr] + bb;
}

// ---------------------------- out[i,j,h] = tanh(rp[i,h] + hx[j,h])  (quirk)
__global__ __launch_bounds__(256)
void output_kernel(const float* __restrict__ rp, const float* __restrict__ hx,
                   float* __restrict__ out) {
  int idx = blockIdx.x * 256 + threadIdx.x;   // B*B*H = 4194304 total
  int h = idx & (Hdim - 1);
  int j = (idx >> 10) & (Bsz - 1);
  int i = idx >> 16;
  out[idx] = fast_tanh(rp[i * Hdim + h] + hx[j * Hdim + h]);
}

// --------------------------------------------------------------- entry point
extern "C" void kernel_launch(void* const* d_in, const int* in_sizes, int n_in,
                              void* d_out, int out_size, void* d_ws, size_t ws_size,
                              hipStream_t stream) {
  const float* hidden = (const float*)d_in[0];
  // d_in[1] aspect, d_in[4] W_v, d_in[5] b_v, d_in[7] w_b: softmax-invariant, skipped
  const float* W_h = (const float*)d_in[2];
  const float* b_h = (const float*)d_in[3];
  const float* w_w = (const float*)d_in[6];
  const float* W_p = (const float*)d_in[8];
  const float* b_p = (const float*)d_in[9];
  const float* W_x = (const float*)d_in[10];
  const float* b_x = (const float*)d_in[11];
  float* out = (float*)d_out;

  // workspace layout (bytes)
  char* ws = (char*)d_ws;
  float*    scores = (float*)(ws + 0);          // 32768 f32
  float*    alpha  = (float*)(ws + 131072);     // 32768 f32
  float*    r      = (float*)(ws + 262144);     // 64*1024 f32
  float*    rp     = (float*)(ws + 524288);     // 64*1024 f32
  float*    hx     = (float*)(ws + 786432);     // 64*1024 f32
  _Float16* wh16   = (_Float16*)(ws + 1048576); // 1M f16
  _Float16* wp16   = (_Float16*)(ws + 3145728); // 1M f16
  _Float16* wx16   = (_Float16*)(ws + 5242880); // 1M f16

  const int NW = Hdim * Hdim;
  cvt_f32_f16_kernel<<<NW / 256, 256, 0, stream>>>(W_h, wh16, NW);
  cvt_f32_f16_kernel<<<NW / 256, 256, 0, stream>>>(W_p, wp16, NW);
  cvt_f32_f16_kernel<<<NW / 256, 256, 0, stream>>>(W_x, wx16, NW);

  scores_kernel<<<Mrows / 32, 256, 0, stream>>>(hidden, wh16, b_h, w_w, scores);
  softmax_kernel<<<Bsz, 512, 0, stream>>>(scores, alpha);
  weighted_sum_kernel<<<dim3(Hdim / 256, Bsz), 256, 0, stream>>>(hidden, alpha, r);

  // rp = r @ W_p.T + b_p          (A rows contiguous, stride H)
  gemm64_wmma_kernel<<<dim3(4, 8), 256, 0, stream>>>(r, (long long)Hdim, wp16, b_p, rp);
  // hx = hidden[:, T-1, :] @ W_x.T + b_x   (A row b at hidden+(b*T+T-1)*H)
  gemm64_wmma_kernel<<<dim3(4, 8), 256, 0, stream>>>(
      hidden + (size_t)(Tseq - 1) * Hdim, (long long)Tseq * Hdim, wx16, b_x, hx);

  output_kernel<<<(Bsz * Bsz * Hdim) / 256, 256, 0, stream>>>(rp, hx, out);
}
